// LovaszSoftmax_89970974916937
// MI455X (gfx1250) — matrix operand order repaired and checked by reading, not compiled, not run on hardware
//
#include <hip/hip_runtime.h>
#include <hip/hip_bf16.h>
#include <stdint.h>

// Problem constants (fixed by the reference).
#define C_CLASSES 19
#define HW_PIX    (512 * 512)
#define BATCH     4
#define NPIX      (BATCH * HW_PIX)        // 1,048,576 elements per class

// Radix sort config: 3 passes x 11 bits covers the 33-bit key exactly.
#define RADIX_BITS 11
#define BINS       (1 << RADIX_BITS)      // 2048
#define STILE      1024                   // sort tile (per scatter wave)
#define SNBLK      (NPIX / STILE)         // 1024 sort blocks, exact

// Loss/scan stage tiling.
#define LTILE      4096
#define LBLK       (NPIX / LTILE)         // 256

#if defined(__HIP_DEVICE_COMPILE__) && \
    __has_builtin(__builtin_amdgcn_global_load_async_to_lds_b64) && \
    __has_builtin(__builtin_amdgcn_s_wait_asynccnt)
#define HAVE_ASYNC_LDS 1
#else
#define HAVE_ASYNC_LDS 0
#endif

typedef __attribute__((ext_vector_type(16))) _Float16 v16h;
typedef __attribute__((ext_vector_type(8)))  float    v8f;
typedef __attribute__((ext_vector_type(2)))  int      v2i;
typedef __attribute__((address_space(1))) v2i gv2i;   // global-AS vector ptr elem
typedef __attribute__((address_space(3))) v2i lv2i;   // LDS-AS vector ptr elem

// ---------------------------------------------------------------------------
// Kernel 1: build sort keys for one class.
// key = (~bits(err) << 1) | gt  -> ascending int sort == descending err sort,
// gt bit rides along as a loss-invariant tie-breaker.
// ---------------------------------------------------------------------------
__global__ void build_keys_kernel(const float* __restrict__ logits,
                                  const int* __restrict__ targets,
                                  uint64_t* __restrict__ keys, int cls) {
  int p = blockIdx.x * blockDim.x + threadIdx.x;
  if (p >= NPIX) return;
  int b  = p / HW_PIX;
  int hw = p - b * HW_PIX;
  const float* base = logits + (size_t)b * C_CLASSES * HW_PIX + hw;

  float l[C_CLASSES];
  float m = -3.4e38f;
#pragma unroll
  for (int c = 0; c < C_CLASSES; ++c) {
    l[c] = base[(size_t)c * HW_PIX];
    m = fmaxf(m, l[c]);
  }
  float s = 0.0f;
#pragma unroll
  for (int c = 0; c < C_CLASSES; ++c) s += expf(l[c] - m);
  float prob = expf(l[cls] - m) / s;

  int   t   = targets[p];
  int   gt  = (t == cls) ? 1 : 0;
  float err = gt ? (1.0f - prob) : prob;          // == |onehot - prob|
  uint32_t inv = ~__float_as_uint(err);           // order-reversing on non-neg floats
  keys[p] = (((uint64_t)inv) << 1) | (uint64_t)gt;
}

// ---------------------------------------------------------------------------
// Radix pass A: per-block digit histogram (digit-major global layout).
// ---------------------------------------------------------------------------
__global__ void hist_kernel(const uint64_t* __restrict__ keys,
                            uint32_t* __restrict__ g_hist, int shift) {
  __shared__ uint32_t h[BINS];
  int tid = threadIdx.x;                          // 256 threads
  for (int i = tid; i < BINS; i += 256) h[i] = 0;
  __syncthreads();
  int base = blockIdx.x * STILE;
  for (int i = tid; i < STILE; i += 256) {
    uint32_t d = (uint32_t)(keys[base + i] >> shift) & (BINS - 1);
    atomicAdd(&h[d], 1u);
  }
  __syncthreads();
  for (int i = tid; i < BINS; i += 256)
    g_hist[(size_t)i * SNBLK + blockIdx.x] = h[i];
}

// ---------------------------------------------------------------------------
// Radix pass B1: per-digit exclusive scan over the SNBLK block counts.
// One block per digit (2048 blocks), 256 threads x 4 elements, LDS tree scan.
// Also writes the digit total.
// ---------------------------------------------------------------------------
__global__ void digit_scan_kernel(uint32_t* __restrict__ g_hist,
                                  uint32_t* __restrict__ g_dtot) {
  __shared__ uint32_t sc[256];
  int d = blockIdx.x;
  int t = threadIdx.x;
  size_t row = (size_t)d * SNBLK;
  uint32_t v[4];
  uint32_t s = 0;
#pragma unroll
  for (int j = 0; j < 4; ++j) { v[j] = g_hist[row + 4 * t + j]; s += v[j]; }
  sc[t] = s;
  __syncthreads();
  for (int off = 1; off < 256; off <<= 1) {       // Hillis-Steele inclusive
    uint32_t x = (t >= off) ? sc[t - off] : 0u;
    __syncthreads();
    sc[t] += x;
    __syncthreads();
  }
  if (t == 255) g_dtot[d] = sc[255];
  uint32_t run = sc[t] - s;                       // exclusive prefix for this thread
#pragma unroll
  for (int j = 0; j < 4; ++j) { uint32_t x = v[j]; g_hist[row + 4 * t + j] = run; run += x; }
}

// ---------------------------------------------------------------------------
// Radix pass B2: exclusive scan of the 2048 digit totals -> digit bases.
// Single block, 256 threads x 8 digits.
// ---------------------------------------------------------------------------
__global__ void base_scan_kernel(const uint32_t* __restrict__ g_dtot,
                                 uint32_t* __restrict__ g_dbase) {
  __shared__ uint32_t sc[256];
  int t = threadIdx.x;
  uint32_t v[8];
  uint32_t s = 0;
#pragma unroll
  for (int j = 0; j < 8; ++j) { v[j] = g_dtot[8 * t + j]; s += v[j]; }
  sc[t] = s;
  __syncthreads();
  for (int off = 1; off < 256; off <<= 1) {
    uint32_t x = (t >= off) ? sc[t - off] : 0u;
    __syncthreads();
    sc[t] += x;
    __syncthreads();
  }
  uint32_t run = sc[t] - s;
#pragma unroll
  for (int j = 0; j < 8; ++j) { uint32_t x = v[j]; g_dbase[8 * t + j] = run; run += x; }
}

// ---------------------------------------------------------------------------
// Radix pass C: stable scatter. One wave32 per 1024-element tile; peer masks
// via 11 ballots, leader lane bumps the LDS running offset for its digit.
// The digit base is folded into the LDS preload (saves a full add-base pass).
// ---------------------------------------------------------------------------
__global__ void scatter_kernel(const uint64_t* __restrict__ src,
                               uint64_t* __restrict__ dst,
                               const uint32_t* __restrict__ g_hist,
                               const uint32_t* __restrict__ g_dbase, int shift) {
  __shared__ uint32_t off[BINS];
  int lane = threadIdx.x;                         // 32 threads
  int blk  = blockIdx.x;
  for (int i = lane; i < BINS; i += 32)
    off[i] = g_hist[(size_t)i * SNBLK + blk] + g_dbase[i];
  __syncthreads();
  int base = blk * STILE;
  for (int chunk = 0; chunk < STILE / 32; ++chunk) {
    if (chunk + 1 < STILE / 32)
      __builtin_prefetch(&src[base + (chunk + 1) * 32 + lane], 0, 3);
    uint64_t key = src[base + chunk * 32 + lane];
    uint32_t dig = (uint32_t)(key >> shift) & (BINS - 1);
    uint32_t peers = 0xFFFFFFFFu;
#pragma unroll
    for (int bit = 0; bit < RADIX_BITS; ++bit) {
      uint32_t vote = (uint32_t)__ballot((dig >> bit) & 1u);
      peers &= ((dig >> bit) & 1u) ? vote : ~vote;
    }
    uint32_t rank   = __popc(peers & ((1u << lane) - 1u));
    int      leader = __ffs(peers) - 1;
    uint32_t prior  = 0;
    if (lane == leader) {
      prior = off[dig];
      off[dig] = prior + __popc(peers);
    }
    prior = __shfl(prior, leader);
    dst[prior + rank] = key;
  }
}

// ---------------------------------------------------------------------------
// Scan stage 1: per-block gt-bit sums over the sorted keys.
// ---------------------------------------------------------------------------
__global__ void bsum_kernel(const uint64_t* __restrict__ keys,
                            uint32_t* __restrict__ bsum) {
  __shared__ uint32_t red[256];
  int tid  = threadIdx.x;
  int base = blockIdx.x * LTILE;
  uint32_t s = 0;
  for (int i = tid; i < LTILE; i += 256) s += (uint32_t)(keys[base + i] & 1ull);
  red[tid] = s;
  __syncthreads();
  for (int o = 128; o > 0; o >>= 1) {
    if (tid < o) red[tid] += red[tid + o];
    __syncthreads();
  }
  if (tid == 0) bsum[blockIdx.x] = red[0];
}

// Scan stage 2: exclusive scan of the 256 block sums (sequential, exact).
__global__ void bscan_kernel(const uint32_t* __restrict__ bsum,
                             uint32_t* __restrict__ bpre,
                             uint32_t* __restrict__ gts) {
  if (threadIdx.x == 0) {
    uint32_t run = 0;
    for (int b = 0; b < LBLK; ++b) { uint32_t v = bsum[b]; bpre[b] = run; run += v; }
    gts[0] = run;
  }
}

// ---------------------------------------------------------------------------
// Scan stage 3 + Lovasz reduction. One wave32 per block; each 16x16 tile's
// inclusive row-prefix of gt bits is computed with v_wmma_f32_16x16x32_f16
// against an upper-triangular ones matrix (exact: 0/1 values, sums <= 256).
// Tiles are staged into LDS with the CDNA5 async Global->LDS path when the
// toolchain exposes it. J_i = 1 - (gts - cs_i)/(gts + i - cs_i).
// ---------------------------------------------------------------------------
__device__ __forceinline__ float jacc(float gts, float cs, float i) {
  return 1.0f - (gts - cs) / (gts + i - cs);
}

__global__ void loss_kernel(const uint64_t* __restrict__ keys,
                            const uint32_t* __restrict__ bpre,
                            const uint32_t* __restrict__ gts_p,
                            float* __restrict__ partial, int cls) {
  __shared__ uint64_t tileK[256];
  __shared__ float rowTot[16];
  __shared__ float excRow[17];

  int lane = threadIdx.x;                         // 32 threads = 1 wave
  int blk  = blockIdx.x;
  bool hiHalf = lane >= 16;
  int  nc  = lane & 15;                           // column (N) for B/D

  float gts   = (float)gts_p[0];
  float carry = (float)bpre[blk];
  float acc   = 0.0f;

  // B: 32x16, rows K<16 upper-triangular ones (B[k][n] = k<=n), rows 16..31 zero.
  // Layout (f16 B): VGPR j -> lanes 0-15 hold K={2j,2j+1}, lanes 16-31 K={2j+16,2j+17}.
  v16h bm;
#pragma unroll
  for (int j = 0; j < 8; ++j) {
    int k0 = hiHalf ? (2 * j + 16) : (2 * j);
    bm[2 * j]     = (_Float16)((!hiHalf && (k0 <= nc))     ? 1.0f : 0.0f);
    bm[2 * j + 1] = (_Float16)((!hiHalf && (k0 + 1 <= nc)) ? 1.0f : 0.0f);
  }

  int tileBase = blk * LTILE;                     // 16 tiles of 256 per block
  for (int tile = 0; tile < 16; ++tile) {
    int tb = tileBase + tile * 256;
#if HAVE_ASYNC_LDS
    {
      uint64_t* kp = const_cast<uint64_t*>(keys);  // builtin takes non-const ptr
#pragma unroll
      for (int j = 0; j < 8; ++j) {
        __builtin_amdgcn_global_load_async_to_lds_b64(
            (gv2i*)&kp[tb + lane * 8 + j],
            (lv2i*)&tileK[lane * 8 + j],
            0, 0);
      }
    }
    __builtin_amdgcn_s_wait_asynccnt(0);
    __syncthreads();
#else
#pragma unroll
    for (int j = 0; j < 8; ++j) tileK[lane * 8 + j] = keys[tb + lane * 8 + j];
    __syncthreads();
#endif

    // A: 16x32 f16, X[m][k] = gt bit of element m*16+k for k<16, zero-pad k>=16.
    // Layout: lanes 0-15 (m=lane): VGPR j K={2j,2j+1} (K=0..15).
    //         lanes 16-31 (m=lane-16): VGPR j K={2j+16,2j+17} (K=16..31).
    v16h am;
    int m = lane & 15;
#pragma unroll
    for (int j = 0; j < 8; ++j) {
      int k0 = hiHalf ? (2 * j + 16) : (2 * j);
      float v0 = 0.0f, v1 = 0.0f;
      if (k0 < 16) {
        v0 = (float)(tileK[m * 16 + k0] & 1ull);
        v1 = (float)(tileK[m * 16 + k0 + 1] & 1ull);
      }
      am[2 * j]     = (_Float16)v0;
      am[2 * j + 1] = (_Float16)v1;
    }

    v8f cz = {};
    v8f d = __builtin_amdgcn_wmma_f32_16x16x32_f16(
        false, am, false, bm, (short)0, cz, false, false);
    // D layout: VGPR i -> lanes 0-15: (M=i, N=lane); lanes 16-31: (M=i+8, N=lane-16).

    // Row totals Y[r][15] live in lane 15 (rows 0-7) and lane 31 (rows 8-15).
    if (lane == 15 || lane == 31) {
      int rb = (lane == 31) ? 8 : 0;
      for (int i = 0; i < 8; ++i) rowTot[rb + i] = d[i];
    }
    __syncthreads();
    if (lane == 0) {
      float run = 0.0f;
      for (int r = 0; r < 16; ++r) { excRow[r] = run; run += rowTot[r]; }
      excRow[16] = run;                           // tile total
    }
    __syncthreads();

#pragma unroll
    for (int i = 0; i < 8; ++i) {
      int r   = hiHalf ? (i + 8) : i;
      int idx = r * 16 + nc;
      uint64_t key = tileK[idx];
      float gt = (float)(key & 1ull);
      float e  = __uint_as_float(~(uint32_t)(key >> 1));
      float cs = carry + excRow[r] + d[i];        // inclusive gt prefix
      float fi = (float)(tb + idx + 1);           // 1-indexed global rank
      float J  = jacc(gts, cs, fi);
      float Jp = (tb + idx == 0) ? 0.0f : jacc(gts, cs - gt, fi - 1.0f);
      acc += e * (J - Jp);
    }
    carry += excRow[16];
    __syncthreads();
  }

  for (int o = 16; o > 0; o >>= 1) acc += __shfl_down(acc, o);
  if (lane == 0) partial[cls * LBLK + blk] = acc;
}

// Deterministic final reduction over 19*256 partials; mean over classes.
__global__ void final_reduce_kernel(const float* __restrict__ partial,
                                    float* __restrict__ out) {
  __shared__ float red[256];
  int tid = threadIdx.x;
  float s = 0.0f;
  for (int i = tid; i < C_CLASSES * LBLK; i += 256) s += partial[i];
  red[tid] = s;
  __syncthreads();
  for (int o = 128; o > 0; o >>= 1) {
    if (tid < o) red[tid] += red[tid + o];
    __syncthreads();
  }
  if (tid == 0) out[0] = red[0] * (1.0f / (float)C_CLASSES);
}

// ---------------------------------------------------------------------------
extern "C" void kernel_launch(void* const* d_in, const int* in_sizes, int n_in,
                              void* d_out, int out_size, void* d_ws, size_t ws_size,
                              hipStream_t stream) {
  (void)in_sizes; (void)n_in; (void)out_size; (void)ws_size;
  const float* logits  = (const float*)d_in[0];
  const int*   targets = (const int*)d_in[1];
  float*       out     = (float*)d_out;

  char* ws = (char*)d_ws;
  uint64_t* keysA  = (uint64_t*)ws;                                   // 8 MB
  uint64_t* keysB  = (uint64_t*)(ws + (size_t)NPIX * 8);              // 8 MB
  uint32_t* g_hist = (uint32_t*)(ws + (size_t)NPIX * 16);             // 8 MB
  char*     tail   = ws + (size_t)NPIX * 16 + (size_t)BINS * SNBLK * 4;
  uint32_t* g_dtot  = (uint32_t*)tail;                                // 8 KB
  uint32_t* g_dbase = g_dtot + BINS;                                  // 8 KB
  uint32_t* g_bsum  = g_dbase + BINS;                                 // 1 KB
  uint32_t* g_bpre  = g_bsum + LBLK;                                  // 1 KB
  uint32_t* g_gts   = g_bpre + LBLK;
  float*    g_part  = (float*)(g_gts + 64);                           // 19*256 floats

  static const int shifts[3] = {0, 11, 22};

  for (int cls = 0; cls < C_CLASSES; ++cls) {
    build_keys_kernel<<<NPIX / 256, 256, 0, stream>>>(logits, targets, keysA, cls);
    for (int pass = 0; pass < 3; ++pass) {
      const uint64_t* src = (pass & 1) ? keysB : keysA;
      uint64_t*       dst = (pass & 1) ? keysA : keysB;
      hist_kernel      <<<SNBLK, 256, 0, stream>>>(src, g_hist, shifts[pass]);
      digit_scan_kernel<<<BINS,  256, 0, stream>>>(g_hist, g_dtot);
      base_scan_kernel <<<1,     256, 0, stream>>>(g_dtot, g_dbase);
      scatter_kernel   <<<SNBLK, 32,  0, stream>>>(src, dst, g_hist, g_dbase,
                                                   shifts[pass]);
    }
    // After passes A->B, B->A, A->B the sorted keys are in keysB.
    bsum_kernel <<<LBLK, 256, 0, stream>>>(keysB, g_bsum);
    bscan_kernel<<<1,    32,  0, stream>>>(g_bsum, g_bpre, g_gts);
    loss_kernel <<<LBLK, 32,  0, stream>>>(keysB, g_bpre, g_gts, g_part, cls);
  }
  final_reduce_kernel<<<1, 256, 0, stream>>>(g_part, out);
}
